// FAGCN_29231547416625
// MI455X (gfx1250) — compile-verified, matrix operand order checked by reading, not run on hardware
//
#include <hip/hip_runtime.h>
#include <math.h>

#define N_NODES 100000
#define N_EDGES 1600000
#define IN_F    500
#define H_F     128
#define C_CLS   40
#define EPSV    0.1f

typedef __attribute__((ext_vector_type(16))) __bf16 bf16x16;
typedef __attribute__((ext_vector_type(8)))  __bf16 bf16x8;
typedef __attribute__((ext_vector_type(8)))  float  f32x8;

// ---------------------------------------------------------------------------
// degree / norm kernels
// ---------------------------------------------------------------------------
__global__ void deg_init_kernel(float* deg, int n) {
    int i = blockIdx.x * blockDim.x + threadIdx.x;
    if (i < n) deg[i] = 1.0f;            // self-loop contribution
}

__global__ void deg_accum_kernel(const int* __restrict__ ei, float* deg, int e) {
    int i = blockIdx.x * blockDim.x + threadIdx.x;
    if (i < e) atomicAdd(&deg[ei[N_EDGES + i]], 1.0f);   // dst row of edge_index
}

__global__ void deg_rsqrt_kernel(float* deg, int n) {
    int i = blockIdx.x * blockDim.x + threadIdx.x;
    if (i < n) deg[i] = rsqrtf(deg[i]);
}

// ---------------------------------------------------------------------------
// WMMA bf16 GEMM:  h[N,128] = x[N,500] @ W[500,128] + b
// block = 256 threads (8 waves); each wave -> 16 rows x 128 cols
// ---------------------------------------------------------------------------
#define GM_BM   128
#define GM_BK   32
#define GM_KP   40      // padded LDS K stride (bf16 units): conflict-free b128

__device__ __forceinline__ bf16x16 frag_from_lds(const __bf16* rowbase, int kb) {
    const bf16x8* p0 = reinterpret_cast<const bf16x8*>(rowbase + kb);
    const bf16x8* p1 = reinterpret_cast<const bf16x8*>(rowbase + kb + 16);
    bf16x8 lo = *p0;
    bf16x8 hi = *p1;
    bf16x16 r;
#pragma unroll
    for (int i = 0; i < 8; ++i) { r[i] = lo[i]; r[i + 8] = hi[i]; }
    return r;
}

__global__ void __launch_bounds__(256)
gemm_in_kernel(const float* __restrict__ x, const float* __restrict__ W,
               const float* __restrict__ b, float* __restrict__ h) {
    __shared__ __align__(16) __bf16 At[GM_BM][GM_KP];
    __shared__ __align__(16) __bf16 Bt[H_F][GM_KP];

    const int tid  = threadIdx.x;
    const int lane = tid & 31;
    const int wave = tid >> 5;                 // 0..7
    const int mblk = blockIdx.x * GM_BM;

    f32x8 acc[8] = {};

    for (int k0 = 0; k0 < IN_F; k0 += GM_BK) {
        // ---- stage A tile (128 rows x 32 cols of x), fp32 -> bf16 ----
        {
            int row  = tid >> 1;               // 2 threads per row
            int cg   = (tid & 1) * 16;
            int grow = mblk + row;
            if (grow >= N_NODES) grow = N_NODES - 1;    // clamp (dup rows ok)
            const float* src = x + (size_t)grow * IN_F;
            if (k0 + GM_BK <= IN_F) {
                const float4* s4 = reinterpret_cast<const float4*>(src + k0 + cg);
#pragma unroll
                for (int u = 0; u < 4; ++u) {
                    float4 v = s4[u];
                    At[row][cg + 4*u + 0] = (__bf16)v.x;
                    At[row][cg + 4*u + 1] = (__bf16)v.y;
                    At[row][cg + 4*u + 2] = (__bf16)v.z;
                    At[row][cg + 4*u + 3] = (__bf16)v.w;
                }
                // prefetch next K tile of x (global_prefetch_b8)
                if (k0 + GM_BK < IN_F)
                    __builtin_prefetch(src + k0 + GM_BK + cg, 0, 0);
            } else {
#pragma unroll
                for (int u = 0; u < 16; ++u) {
                    int kk = k0 + cg + u;
                    At[row][cg + u] = (__bf16)((kk < IN_F) ? src[kk] : 0.0f);
                }
            }
        }
        // ---- stage B tile (32 K-rows x 128 cols of W), transposed -> bf16 ----
        {
            int kr = tid >> 3;                 // 0..31
            int cg = (tid & 7) * 16;
            int kk = k0 + kr;
            if (kk < IN_F) {
                const float4* s4 = reinterpret_cast<const float4*>(W + (size_t)kk * H_F + cg);
#pragma unroll
                for (int u = 0; u < 4; ++u) {
                    float4 v = s4[u];
                    Bt[cg + 4*u + 0][kr] = (__bf16)v.x;
                    Bt[cg + 4*u + 1][kr] = (__bf16)v.y;
                    Bt[cg + 4*u + 2][kr] = (__bf16)v.z;
                    Bt[cg + 4*u + 3][kr] = (__bf16)v.w;
                }
            } else {
#pragma unroll
                for (int u = 0; u < 16; ++u) Bt[cg + u][kr] = (__bf16)0.0f;
            }
        }
        __syncthreads();

        // ---- fragments + 8 WMMAs (full EXEC, no divergence here) ----
        const int r   = lane & 15;
        const int kb  = (lane < 16) ? 0 : 8;   // ISA 16-bit A/B lane layout
        bf16x16 afrag = frag_from_lds(&At[wave * 16 + r][0], kb);
#pragma unroll
        for (int j = 0; j < 8; ++j) {
            bf16x16 bfrag = frag_from_lds(&Bt[j * 16 + r][0], kb);
            acc[j] = __builtin_amdgcn_wmma_f32_16x16x32_bf16(
                false, afrag, false, bfrag, (short)0, acc[j], false, false);
        }
        __syncthreads();
    }

    // ---- epilogue: + bias, guarded stores ----
    const int r16  = lane & 15;
    const int mofs = wave * 16 + ((lane >> 4) << 3);   // upper half-wave -> +8
#pragma unroll
    for (int j = 0; j < 8; ++j) {
        int   col  = j * 16 + r16;
        float bias = b[col];
#pragma unroll
        for (int rr = 0; rr < 8; ++rr) {
            int m = mblk + mofs + rr;
            if (m < N_NODES) h[(size_t)m * H_F + col] = acc[j][rr] + bias;
        }
    }
}

// ---------------------------------------------------------------------------
// per-node attention dots: al[i] = h[i]·attl, ar[i] = h[i]·attr  (wave/node)
// ---------------------------------------------------------------------------
__global__ void att_dot_kernel(const float* __restrict__ h,
                               const float* __restrict__ attl,
                               const float* __restrict__ attr,
                               float* __restrict__ al, float* __restrict__ ar) {
    int wid  = (blockIdx.x * blockDim.x + threadIdx.x) >> 5;   // node
    int lane = threadIdx.x & 31;
    if (wid >= N_NODES) return;
    float4 hv = reinterpret_cast<const float4*>(h + (size_t)wid * H_F)[lane];
    float4 lv = reinterpret_cast<const float4*>(attl)[lane];
    float4 rv = reinterpret_cast<const float4*>(attr)[lane];
    float sl = hv.x*lv.x + hv.y*lv.y + hv.z*lv.z + hv.w*lv.w;
    float sr = hv.x*rv.x + hv.y*rv.y + hv.z*rv.z + hv.w*rv.w;
#pragma unroll
    for (int m = 16; m > 0; m >>= 1) {
        sl += __shfl_xor(sl, m, 32);
        sr += __shfl_xor(sr, m, 32);
    }
    if (lane == 0) { al[wid] = sl; ar[wid] = sr; }
}

// ---------------------------------------------------------------------------
// agg = EPS * h0   (residual init before scatter)
// ---------------------------------------------------------------------------
__global__ void scale_init_kernel(const float* __restrict__ h0,
                                  float* __restrict__ dst, int n4) {
    int i = blockIdx.x * blockDim.x + threadIdx.x;
    if (i < n4) {
        float4 v = reinterpret_cast<const float4*>(h0)[i];
        float4 o; o.x = EPSV*v.x; o.y = EPSV*v.y; o.z = EPSV*v.z; o.w = EPSV*v.w;
        reinterpret_cast<float4*>(dst)[i] = o;
    }
}

// ---------------------------------------------------------------------------
// edge scatter: one wave32 per edge (incl. self-loops), 4 floats/lane,
// agg[dst] += tanh(al[src]+ar[dst]) * dinv[src]*dinv[dst] * feat[src]
// ---------------------------------------------------------------------------
__global__ void __launch_bounds__(256)
edge_scatter_kernel(const int* __restrict__ ei, const float* __restrict__ feat,
                    const float* __restrict__ al, const float* __restrict__ ar,
                    const float* __restrict__ dinv, float* __restrict__ agg) {
    int wid  = (blockIdx.x * blockDim.x + threadIdx.x) >> 5;   // edge id
    int lane = threadIdx.x & 31;
    int s, d;
    if (wid < N_EDGES) { s = ei[wid]; d = ei[N_EDGES + wid]; }
    else               { s = d = wid - N_EDGES; }              // self-loop
    float coeff = tanhf(al[s] + ar[d]) * dinv[s] * dinv[d];
    float4 f = reinterpret_cast<const float4*>(feat + (size_t)s * H_F)[lane];
    float* dp = agg + (size_t)d * H_F + lane * 4;
    atomicAdd(dp + 0, coeff * f.x);
    atomicAdd(dp + 1, coeff * f.y);
    atomicAdd(dp + 2, coeff * f.z);
    atomicAdd(dp + 3, coeff * f.w);
}

__global__ void relu_kernel(float* __restrict__ h, int n4) {
    int i = blockIdx.x * blockDim.x + threadIdx.x;
    if (i < n4) {
        float4 v = reinterpret_cast<float4*>(h)[i];
        v.x = fmaxf(v.x, 0.f); v.y = fmaxf(v.y, 0.f);
        v.z = fmaxf(v.z, 0.f); v.w = fmaxf(v.w, 0.f);
        reinterpret_cast<float4*>(h)[i] = v;
    }
}

// ---------------------------------------------------------------------------
// classifier + softmax + argmax (thread per node, W_cls in LDS -> broadcast)
// ---------------------------------------------------------------------------
__global__ void __launch_bounds__(256)
cls_kernel(const float* __restrict__ emb, const float* __restrict__ Wc,
           const float* __restrict__ bc, float* __restrict__ out_logits,
           float* __restrict__ out_soft, float* __restrict__ out_hard) {
    __shared__ float Ws[H_F][C_CLS];
    __shared__ float bs[C_CLS];
    for (int idx = threadIdx.x; idx < H_F * C_CLS; idx += blockDim.x)
        Ws[idx / C_CLS][idx % C_CLS] = Wc[idx];
    if (threadIdx.x < C_CLS) bs[threadIdx.x] = bc[threadIdx.x];
    __syncthreads();

    int i = blockIdx.x * blockDim.x + threadIdx.x;
    if (i >= N_NODES) return;

    float acc[C_CLS];
#pragma unroll
    for (int c = 0; c < C_CLS; ++c) acc[c] = bs[c];

    const float4* e4 = reinterpret_cast<const float4*>(emb + (size_t)i * H_F);
    for (int k = 0; k < H_F / 4; ++k) {
        float4 e = e4[k];
#pragma unroll
        for (int c = 0; c < C_CLS; ++c)
            acc[c] += e.x * Ws[4*k+0][c] + e.y * Ws[4*k+1][c]
                    + e.z * Ws[4*k+2][c] + e.w * Ws[4*k+3][c];
    }

    float maxv = acc[0]; int bi = 0;
#pragma unroll
    for (int c = 1; c < C_CLS; ++c)
        if (acc[c] > maxv) { maxv = acc[c]; bi = c; }
    float sum = 0.f;
    float ex[C_CLS];
#pragma unroll
    for (int c = 0; c < C_CLS; ++c) { ex[c] = __expf(acc[c] - maxv); sum += ex[c]; }
    float inv = 1.0f / sum;

    float* lg = out_logits + (size_t)i * C_CLS;
    float* sf = out_soft   + (size_t)i * C_CLS;
#pragma unroll
    for (int c = 0; c < C_CLS; ++c) { lg[c] = acc[c]; sf[c] = ex[c] * inv; }
    out_hard[i] = (float)bi;
}

// ---------------------------------------------------------------------------
// launch
// ---------------------------------------------------------------------------
extern "C" void kernel_launch(void* const* d_in, const int* in_sizes, int n_in,
                              void* d_out, int out_size, void* d_ws, size_t ws_size,
                              hipStream_t stream) {
    const float* x     = (const float*)d_in[0];
    const int*   ei    = (const int*)  d_in[1];
    const float* W_in  = (const float*)d_in[2];
    const float* b_in  = (const float*)d_in[3];
    const float* attl1 = (const float*)d_in[4];
    const float* attr1 = (const float*)d_in[5];
    const float* attl2 = (const float*)d_in[6];
    const float* attr2 = (const float*)d_in[7];
    const float* W_cls = (const float*)d_in[8];
    const float* b_cls = (const float*)d_in[9];

    float* out    = (float*)d_out;
    float* logits = out;                                          // [N,C]
    float* emb    = out + (size_t)N_NODES * C_CLS;                // [N,H]
    float* soft   = emb + (size_t)N_NODES * H_F;                  // [N,C]
    float* hard   = soft + (size_t)N_NODES * C_CLS;               // [N]

    float* ws   = (float*)d_ws;
    float* dinv = ws;                                             // N
    float* al   = dinv + N_NODES;                                 // N
    float* ar   = al + N_NODES;                                   // N
    float* h0   = ar + N_NODES;                                   // N*H
    float* h1   = h0 + (size_t)N_NODES * H_F;                     // N*H

    const int nh4 = (N_NODES * H_F) / 4;                          // 3,200,000
    const int edge_blocks = ((N_EDGES + N_NODES) * 32) / 256;     // 212,500
    const int node_wave_blocks = (N_NODES * 32) / 256;            // 12,500

    // degrees / norm
    deg_init_kernel <<<(N_NODES + 255) / 256, 256, 0, stream>>>(dinv, N_NODES);
    deg_accum_kernel<<<(N_EDGES + 255) / 256, 256, 0, stream>>>(ei, dinv, N_EDGES);
    deg_rsqrt_kernel<<<(N_NODES + 255) / 256, 256, 0, stream>>>(dinv, N_NODES);

    // input projection (WMMA bf16)
    gemm_in_kernel<<<(N_NODES + GM_BM - 1) / GM_BM, 256, 0, stream>>>(x, W_in, b_in, h0);

    // FAConv layer 1
    att_dot_kernel   <<<node_wave_blocks, 256, 0, stream>>>(h0, attl1, attr1, al, ar);
    scale_init_kernel<<<(nh4 + 255) / 256, 256, 0, stream>>>(h0, h1, nh4);
    edge_scatter_kernel<<<edge_blocks, 256, 0, stream>>>(ei, h0, al, ar, dinv, h1);
    relu_kernel      <<<(nh4 + 255) / 256, 256, 0, stream>>>(h1, nh4);

    // FAConv layer 2 (writes embedding directly into d_out slot)
    att_dot_kernel   <<<node_wave_blocks, 256, 0, stream>>>(h1, attl2, attr2, al, ar);
    scale_init_kernel<<<(nh4 + 255) / 256, 256, 0, stream>>>(h0, emb, nh4);
    edge_scatter_kernel<<<edge_blocks, 256, 0, stream>>>(ei, h1, al, ar, dinv, emb);

    // classifier + softmax + argmax
    cls_kernel<<<(N_NODES + 255) / 256, 256, 0, stream>>>(emb, W_cls, b_cls,
                                                          logits, soft, hard);
}
